// MultiHeadSWAttention_9835475107988
// MI455X (gfx1250) — matrix operand order repaired and checked by reading, not compiled, not run on hardware
//
#include <hip/hip_runtime.h>
#include <cstdint>

// ---------------------------------------------------------------------------
// Sliding-window MHA for MI455X (gfx1250, wave32, WMMA + TDM).
// Fixed reference dims: B=2, L=2048, D_MODEL=1024, H=16, d_k=d_v=64.
// Pipeline (all bf16 WMMA, f32 accumulate):
//   0) zero-fill attention output
//   1) cvt: stream-convert q,k,v,Wq,Wk,Wv,Wo to bf16 (one pass, BW-trivial)
//   2) proj_qkv: TDM-staged bf16 GEMM -> Q,K (B,H,L,64) ; V^T (B,H,64,L)
//   3) attn: per-wave 16-query tile, 2-pass online softmax over key window,
//      writes attention probs (f32, the 537MB roofline term) + O bf16
//   4) out_proj: TDM-staged bf16 GEMM, Y = O @ Wo^T + residual(q)
//   5) layernorm -> d_out x
// Workspace (56 MiB, with liveness reuse):
//   [0,8M)   Xq bf16   -> reused as Obf after proj
//   [8,24M)  Xk,Xv bf16-> reused as Y (f32) after proj
//   [24,32M) Wbf (Wq|Wk|Wv|Wo, 2MiB each)
//   [32,56M) Qbf | Kbf | Vtbf
// ---------------------------------------------------------------------------

#define B_SZ    2
#define L_SEQ   2048
#define D_MODEL 1024
#define H_N     16
#define D_HEAD  64
#define ROWS    (B_SZ * L_SEQ)   // 4096
#define NEGBIG  (-1.0e9f)

typedef __attribute__((ext_vector_type(16))) __bf16   v16bf;
typedef __attribute__((ext_vector_type(2)))  __bf16   v2bf;
typedef __attribute__((ext_vector_type(8)))  float    v8f;
typedef __attribute__((ext_vector_type(8)))  unsigned v8u;
typedef __attribute__((ext_vector_type(4)))  unsigned v4u;
typedef __attribute__((ext_vector_type(4)))  int      v4i;
typedef __attribute__((ext_vector_type(8)))  int      v8i;

static __device__ __forceinline__ unsigned short f2bf(float f) {
  return __builtin_bit_cast(unsigned short, (__bf16)f);
}
// packed f32x2 -> bf16x2 (single v_cvt_pk_bf16_f32)
static __device__ __forceinline__ unsigned pk2(float a, float b) {
  v2bf t;
  t[0] = (__bf16)a;
  t[1] = (__bf16)b;
  return __builtin_bit_cast(unsigned, t);
}
static __device__ __forceinline__ v8f v8f_zero() {
  v8f z;
#pragma unroll
  for (int i = 0; i < 8; i++) z[i] = 0.0f;
  return z;
}
static __device__ __forceinline__ v8f wmma_bf16(v16bf a, v16bf b, v8f c) {
  return __builtin_amdgcn_wmma_f32_16x16x32_bf16(false, a, false, b,
                                                 (short)0, c, false, false);
}
static __device__ __forceinline__ float redmax16(float v) {
#pragma unroll
  for (int o = 8; o > 0; o >>= 1) v = fmaxf(v, __shfl_xor(v, o, 16));
  return v;
}
static __device__ __forceinline__ float redadd16(float v) {
#pragma unroll
  for (int o = 8; o > 0; o >>= 1) v += __shfl_xor(v, o, 16);
  return v;
}
static __device__ __forceinline__ float adj_score(float raw, int qg, int key,
                                                  float mq, int W2,
                                                  const float* bt) {
  int rel = qg - key;
  int ar = rel < 0 ? -rel : rel;
  if (ar <= W2 && mq != 0.0f) return raw * 0.125f - bt[ar];
  return NEGBIG;
}

// ---------------------------------------------------------------------------
// TDM: async 2D bf16 tile load global -> LDS (CDNA5 ISA).
// D# group0: count=1 | lds_addr | global_addr(57b) | type=2("image")
// D# group1: data_size=1(2B) | tensor_dim0/1 | tile_dim0/1 | dim0 stride
// ---------------------------------------------------------------------------
static __device__ __forceinline__ void tdm_load_2d_bf16(
    unsigned lds_off, const void* gaddr, unsigned tile_d0, unsigned tile_d1,
    unsigned tensor_d0, unsigned tensor_d1, unsigned row_stride_elems) {
  unsigned long long ga = (unsigned long long)(uintptr_t)gaddr;
  v4u g0;
  g0[0] = 1u;
  g0[1] = lds_off;
  g0[2] = (unsigned)(ga & 0xFFFFFFFFu);
  g0[3] = (unsigned)((ga >> 32) & 0x01FFFFFFu) | 0x80000000u;
  v8i g1;
  g1[0] = (int)(1u << 16);                               // data_size = 2B
  g1[1] = (int)((tensor_d0 & 0xFFFFu) << 16);
  g1[2] = (int)(((tensor_d0 >> 16) & 0xFFFFu) | ((tensor_d1 & 0xFFFFu) << 16));
  g1[3] = (int)(((tensor_d1 >> 16) & 0xFFFFu) | (tile_d0 << 16));
  g1[4] = (int)(tile_d1 & 0xFFFFu);
  g1[5] = (int)row_stride_elems;
  g1[6] = 0;
  g1[7] = 0;
  v4i z4 = {0, 0, 0, 0};
#if defined(__clang_major__) && (__clang_major__ >= 23)
  v8i z8 = {0, 0, 0, 0, 0, 0, 0, 0};
  __builtin_amdgcn_tensor_load_to_lds(g0, g1, z4, z4, z8, 0);
#else
  __builtin_amdgcn_tensor_load_to_lds(g0, g1, z4, z4, 0);
#endif
}

// ---------------------------------------------------------------------------
// 0) zero-fill attention output
// ---------------------------------------------------------------------------
__global__ void zero_f4_kernel(float4* __restrict__ p, long long n4) {
  long long i = (long long)blockIdx.x * blockDim.x + threadIdx.x;
  const long long stride = (long long)gridDim.x * blockDim.x;
  const float4 z = make_float4(0.f, 0.f, 0.f, 0.f);
  for (; i < n4; i += stride) p[i] = z;
}

// ---------------------------------------------------------------------------
// 1) streaming f32 -> bf16 convert of all GEMM operands (grid.y = segment)
// ---------------------------------------------------------------------------
__global__ __launch_bounds__(256) void cvt_bf16_kernel(
    const float* __restrict__ q, const float* __restrict__ k,
    const float* __restrict__ v, const float* __restrict__ Wq,
    const float* __restrict__ Wk, const float* __restrict__ Wv,
    const float* __restrict__ Wo, unsigned* __restrict__ Xq,
    unsigned* __restrict__ Xk, unsigned* __restrict__ Xv,
    unsigned* __restrict__ Wb) {
  const long long NX4 = (long long)ROWS * D_MODEL / 4;
  const long long NW4 = (long long)D_MODEL * D_MODEL / 4;
  const long long WSTRIDE2 = (long long)D_MODEL * D_MODEL / 2;  // uints per W
  const float* src;
  unsigned* dst;
  long long n4;
  switch (blockIdx.y) {
    case 0: src = q;  dst = Xq;                n4 = NX4; break;
    case 1: src = k;  dst = Xk;                n4 = NX4; break;
    case 2: src = v;  dst = Xv;                n4 = NX4; break;
    case 3: src = Wq; dst = Wb + 0 * WSTRIDE2; n4 = NW4; break;
    case 4: src = Wk; dst = Wb + 1 * WSTRIDE2; n4 = NW4; break;
    case 5: src = Wv; dst = Wb + 2 * WSTRIDE2; n4 = NW4; break;
    default: src = Wo; dst = Wb + 3 * WSTRIDE2; n4 = NW4; break;
  }
  long long i = (long long)blockIdx.x * blockDim.x + threadIdx.x;
  const long long stride = (long long)gridDim.x * blockDim.x;
  for (; i < n4; i += stride) {
    float4 x = ((const float4*)src)[i];
    if (i + stride < n4) __builtin_prefetch(&((const float4*)src)[i + stride], 0, 1);
    uint2 o;
    o.x = pk2(x.x, x.y);
    o.y = pk2(x.z, x.w);
    ((uint2*)dst)[i] = o;
  }
}

// ---------------------------------------------------------------------------
// 2) QKV projection (all-bf16, both tiles staged by TDM)
//    Y[m,n] = sum_k X[m,k] * W[n,k]; block = 8 waves, tile M=128 N=64 K=32
// ---------------------------------------------------------------------------
__global__ __launch_bounds__(256) void proj_qkv_kernel(
    const unsigned short* __restrict__ Xqbf,
    const unsigned short* __restrict__ Xkbf,
    const unsigned short* __restrict__ Xvbf,
    const unsigned short* __restrict__ Wbf,
    unsigned short* __restrict__ Qbf, unsigned short* __restrict__ Kbf,
    unsigned short* __restrict__ Vtbf) {
  const int which = blockIdx.z;
  const unsigned short* __restrict__ X =
      which == 0 ? Xqbf : (which == 1 ? Xkbf : Xvbf);
  const unsigned short* __restrict__ W =
      Wbf + (size_t)which * D_MODEL * D_MODEL;

  __shared__ unsigned Xs[128 * 16];  // 128 rows x 32 bf16 (TDM tile)
  __shared__ unsigned Ws[64 * 16];   // 64 rows  x 32 bf16 (TDM tile)

  const int tid = threadIdx.x;
  const int lane = tid & 31, wave = tid >> 5;
  const int half = lane >> 4, l16 = lane & 15;
  const int mbase = blockIdx.x * 128, nbase = blockIdx.y * 64;
  const unsigned xs_lds = (unsigned)(uintptr_t)&Xs[0];
  const unsigned ws_lds = (unsigned)(uintptr_t)&Ws[0];

  v8f acc[4];
#pragma unroll
  for (int n = 0; n < 4; n++) acc[n] = v8f_zero();

  for (int kk = 0; kk < D_MODEL; kk += 32) {
    if (wave == 0) {
      tdm_load_2d_bf16(xs_lds, X + (size_t)mbase * D_MODEL + kk,
                       32, 128, D_MODEL, ROWS, D_MODEL);
      tdm_load_2d_bf16(ws_lds, W + (size_t)nbase * D_MODEL + kk,
                       32, 64, D_MODEL, D_MODEL, D_MODEL);
      __builtin_amdgcn_s_wait_tensorcnt(0);
    }
    __syncthreads();

    v8u au;
#pragma unroll
    for (int v = 0; v < 8; v++)
      au[v] = Xs[(wave * 16 + l16) * 16 + half * 8 + v];
    v16bf a = __builtin_bit_cast(v16bf, au);

#pragma unroll
    for (int n = 0; n < 4; n++) {
      v8u bu;
#pragma unroll
      for (int v = 0; v < 8; v++)
        bu[v] = Ws[(n * 16 + l16) * 16 + half * 8 + v];
      acc[n] = wmma_bf16(a, __builtin_bit_cast(v16bf, bu), acc[n]);
    }
    // drain LDS reads before next iteration's TDM overwrites the tiles
    asm volatile("s_wait_dscnt 0x0" ::: "memory");
    __syncthreads();
  }

  unsigned short* __restrict__ out01 = (which == 0) ? Qbf : Kbf;
#pragma unroll
  for (int n = 0; n < 4; n++) {
#pragma unroll
    for (int r = 0; r < 8; r++) {
      int m = mbase + wave * 16 + r + 8 * half;
      int cg = nbase + n * 16 + l16;
      int b = m >> 11, l = m & (L_SEQ - 1);
      int h = cg >> 6, d = cg & 63;
      unsigned short bv = f2bf(acc[n][r]);
      if (which < 2)
        out01[(((size_t)(b * H_N + h) * L_SEQ) + l) * D_HEAD + d] = bv;
      else
        Vtbf[(((size_t)(b * H_N + h) * D_HEAD) + d) * L_SEQ + l] = bv;
    }
  }
}

// ---------------------------------------------------------------------------
// 3) attention: one wave per (b,h,16-query tile); 8 waves / block
// ---------------------------------------------------------------------------
__global__ __launch_bounds__(256) void attn_kernel(
    const unsigned short* __restrict__ Qbf, const unsigned short* __restrict__ Kbf,
    const unsigned short* __restrict__ Vtbf, const float* __restrict__ maskp,
    const float* __restrict__ sf_p, const float* __restrict__ tau_p,
    const int* __restrict__ wsz_p, float* __restrict__ att,
    unsigned short* __restrict__ Obf) {
  __shared__ float bias_tab[2049];
  __shared__ unsigned short Ps[8][16][32];

  const int tid = threadIdx.x, lane = tid & 31, wave = tid >> 5;
  const int half = lane >> 4, l16 = lane & 15;

  const float sf = sf_p[0], tau = tau_p[0];
  const int W2 = wsz_p[0] >> 1;
  const float inv = 1.0f / (fabsf(sf) * tau);
  for (int i = tid; i < 2049; i += 256) bias_tab[i] = __expf(-(float)i * inv);
  __syncthreads();

  const int gw = blockIdx.x * 8 + wave;
  const int qt = gw & 127;
  const int h = (gw >> 7) & 15;
  const int b = gw >> 11;
  const int qlo = qt * 16;
  const size_t bh = (size_t)(b * H_N + h);

  const unsigned* __restrict__ Qrow =
      (const unsigned*)(Qbf + (bh * L_SEQ + qlo + l16) * D_HEAD);
  v8u a0u, a1u;
#pragma unroll
  for (int v = 0; v < 8; v++) {
    int cp = half * 8 + v;
    a0u[v] = Qrow[cp];
    a1u[v] = Qrow[cp + 16];
  }
  const v16bf aq0 = __builtin_bit_cast(v16bf, a0u);
  const v16bf aq1 = __builtin_bit_cast(v16bf, a1u);

  float maskq[8], rmax[8], rsum[8];
#pragma unroll
  for (int r = 0; r < 8; r++) {
    int qg = qlo + r + 8 * half;
    maskq[r] = maskp[b * L_SEQ + qg];
    rmax[r] = -3.0e38f;
    rsum[r] = 0.0f;
  }

  int j0 = qlo - W2; if (j0 < 0) j0 = 0; j0 &= ~31;
  int j1 = qlo + 16 + W2; if (j1 > L_SEQ) j1 = L_SEQ;
  j1 = (j1 + 31) & ~31; if (j1 > L_SEQ) j1 = L_SEQ;

  const unsigned* __restrict__ Kbase = (const unsigned*)(Kbf + bh * (size_t)L_SEQ * D_HEAD);
  const unsigned* __restrict__ Vbase = (const unsigned*)(Vtbf + bh * (size_t)D_HEAD * L_SEQ);

  // ---- pass 1: online row max + sum of exp
  for (int j = j0; j < j1; j += 32) {
    v8f s0 = v8f_zero(), s1 = v8f_zero();
    {
      const unsigned* kr = Kbase + (size_t)(j + l16) * 32;
      v8u b0, b1;
#pragma unroll
      for (int v = 0; v < 8; v++) { int ki = half * 8 + v; b0[v] = kr[ki]; b1[v] = kr[ki + 16]; }
      s0 = wmma_bf16(aq0, __builtin_bit_cast(v16bf, b0), s0);
      s0 = wmma_bf16(aq1, __builtin_bit_cast(v16bf, b1), s0);
    }
    {
      const unsigned* kr = Kbase + (size_t)(j + 16 + l16) * 32;
      v8u b0, b1;
#pragma unroll
      for (int v = 0; v < 8; v++) { int ki = half * 8 + v; b0[v] = kr[ki]; b1[v] = kr[ki + 16]; }
      s1 = wmma_bf16(aq0, __builtin_bit_cast(v16bf, b0), s1);
      s1 = wmma_bf16(aq1, __builtin_bit_cast(v16bf, b1), s1);
    }
#pragma unroll
    for (int r = 0; r < 8; r++) {
      int qg = qlo + r + 8 * half;
      float v0 = adj_score(s0[r], qg, j + l16, maskq[r], W2, bias_tab);
      float v1 = adj_score(s1[r], qg, j + 16 + l16, maskq[r], W2, bias_tab);
      float nm = fmaxf(rmax[r], redmax16(fmaxf(v0, v1)));
      float e = redadd16(__expf(v0 - nm) + __expf(v1 - nm));
      rsum[r] = rsum[r] * __expf(rmax[r] - nm) + e;
      rmax[r] = nm;
    }
  }
  float rinv[8];
#pragma unroll
  for (int r = 0; r < 8; r++) rinv[r] = 1.0f / rsum[r];

  v8f o[4];
#pragma unroll
  for (int n = 0; n < 4; n++) o[n] = v8f_zero();

  // ---- pass 2: probs -> global attention + O accumulation
  for (int j = j0; j < j1; j += 32) {
    v8f s0 = v8f_zero(), s1 = v8f_zero();
    {
      const unsigned* kr = Kbase + (size_t)(j + l16) * 32;
      v8u b0, b1;
#pragma unroll
      for (int v = 0; v < 8; v++) { int ki = half * 8 + v; b0[v] = kr[ki]; b1[v] = kr[ki + 16]; }
      s0 = wmma_bf16(aq0, __builtin_bit_cast(v16bf, b0), s0);
      s0 = wmma_bf16(aq1, __builtin_bit_cast(v16bf, b1), s0);
    }
    {
      const unsigned* kr = Kbase + (size_t)(j + 16 + l16) * 32;
      v8u b0, b1;
#pragma unroll
      for (int v = 0; v < 8; v++) { int ki = half * 8 + v; b0[v] = kr[ki]; b1[v] = kr[ki + 16]; }
      s1 = wmma_bf16(aq0, __builtin_bit_cast(v16bf, b0), s1);
      s1 = wmma_bf16(aq1, __builtin_bit_cast(v16bf, b1), s1);
    }
#pragma unroll
    for (int r = 0; r < 8; r++) {
      int qg = qlo + r + 8 * half;
      float v0 = adj_score(s0[r], qg, j + l16, maskq[r], W2, bias_tab);
      float v1 = adj_score(s1[r], qg, j + 16 + l16, maskq[r], W2, bias_tab);
      float p0 = (v0 > -0.9e9f) ? __expf(v0 - rmax[r]) * rinv[r] : 0.0f;
      float p1 = (v1 > -0.9e9f) ? __expf(v1 - rmax[r]) * rinv[r] : 0.0f;
      size_t arow = (bh * L_SEQ + qg) * (size_t)L_SEQ;
      att[arow + j + l16]      = p0;
      att[arow + j + 16 + l16] = p1;
      Ps[wave][r + 8 * half][l16]      = f2bf(p0);
      Ps[wave][r + 8 * half][l16 + 16] = f2bf(p1);
    }
    asm volatile("s_wait_dscnt 0x0" ::: "memory");

    const unsigned* pr = (const unsigned*)&Ps[wave][l16][0];
    v8u pu;
#pragma unroll
    for (int v = 0; v < 8; v++)
      pu[v] = pr[half * 8 + v];
    v16bf ap = __builtin_bit_cast(v16bf, pu);

#pragma unroll
    for (int n = 0; n < 4; n++) {
      const unsigned* vr = Vbase + (size_t)(n * 16 + l16) * (L_SEQ / 2) + (j >> 1);
      v8u bu;
#pragma unroll
      for (int v = 0; v < 8; v++) bu[v] = vr[half * 8 + v];
      o[n] = wmma_bf16(ap, __builtin_bit_cast(v16bf, bu), o[n]);
    }
  }

#pragma unroll
  for (int n = 0; n < 4; n++) {
#pragma unroll
    for (int r = 0; r < 8; r++) {
      int qg = qlo + r + 8 * half;
      Obf[((size_t)b * L_SEQ + qg) * D_MODEL + h * D_HEAD + n * 16 + l16] =
          f2bf(o[n][r]);
    }
  }
}

// ---------------------------------------------------------------------------
// 4) output projection: Y = O @ Wo^T + residual(q); both tiles via TDM
// ---------------------------------------------------------------------------
__global__ __launch_bounds__(256) void out_proj_kernel(
    const unsigned short* __restrict__ Obf,
    const unsigned short* __restrict__ Wobf,
    const float* __restrict__ resid, float* __restrict__ Y) {
  __shared__ unsigned Xs[128 * 16];
  __shared__ unsigned Ws[64 * 16];
  const int tid = threadIdx.x, lane = tid & 31, wave = tid >> 5;
  const int half = lane >> 4, l16 = lane & 15;
  const int mbase = blockIdx.x * 128, nbase = blockIdx.y * 64;
  const unsigned xs_lds = (unsigned)(uintptr_t)&Xs[0];
  const unsigned ws_lds = (unsigned)(uintptr_t)&Ws[0];

  v8f acc[4];
#pragma unroll
  for (int n = 0; n < 4; n++) acc[n] = v8f_zero();

  for (int kk = 0; kk < D_MODEL; kk += 32) {
    if (wave == 0) {
      tdm_load_2d_bf16(xs_lds, Obf + (size_t)mbase * D_MODEL + kk,
                       32, 128, D_MODEL, ROWS, D_MODEL);
      tdm_load_2d_bf16(ws_lds, Wobf + (size_t)nbase * D_MODEL + kk,
                       32, 64, D_MODEL, D_MODEL, D_MODEL);
      __builtin_amdgcn_s_wait_tensorcnt(0);
    }
    __syncthreads();

    v8u au;
#pragma unroll
    for (int v = 0; v < 8; v++)
      au[v] = Xs[(wave * 16 + l16) * 16 + half * 8 + v];
    v16bf a = __builtin_bit_cast(v16bf, au);

#pragma unroll
    for (int n = 0; n < 4; n++) {
      v8u bu;
#pragma unroll
      for (int v = 0; v < 8; v++) bu[v] = Ws[(n * 16 + l16) * 16 + half * 8 + v];
      acc[n] = wmma_bf16(a, __builtin_bit_cast(v16bf, bu), acc[n]);
    }
    asm volatile("s_wait_dscnt 0x0" ::: "memory");
    __syncthreads();
  }

#pragma unroll
  for (int n = 0; n < 4; n++) {
#pragma unroll
    for (int r = 0; r < 8; r++) {
      int m = mbase + wave * 16 + r + 8 * half;
      int cg = nbase + n * 16 + l16;
      Y[(size_t)m * D_MODEL + cg] = acc[n][r] + resid[(size_t)m * D_MODEL + cg];
    }
  }
}

// ---------------------------------------------------------------------------
// 5) LayerNorm (eps=1e-6), one block per row
// ---------------------------------------------------------------------------
__global__ __launch_bounds__(256) void ln_kernel(const float* __restrict__ Y,
                                                 const float* __restrict__ lnw,
                                                 const float* __restrict__ lnb,
                                                 float* __restrict__ out) {
  __shared__ float red[256];
  const int row = blockIdx.x, tid = threadIdx.x;
  const float* y = Y + (size_t)row * D_MODEL;
  float s = 0.f, s2 = 0.f;
  for (int c = tid; c < D_MODEL; c += 256) { float x = y[c]; s += x; s2 += x * x; }
  red[tid] = s; __syncthreads();
#pragma unroll
  for (int o = 128; o > 0; o >>= 1) { if (tid < o) red[tid] += red[tid + o]; __syncthreads(); }
  const float mu = red[0] * (1.0f / D_MODEL);
  __syncthreads();
  red[tid] = s2; __syncthreads();
#pragma unroll
  for (int o = 128; o > 0; o >>= 1) { if (tid < o) red[tid] += red[tid + o]; __syncthreads(); }
  const float var = red[0] * (1.0f / D_MODEL) - mu * mu;
  const float rs = rsqrtf(var + 1e-6f);
  for (int c = tid; c < D_MODEL; c += 256)
    out[(size_t)row * D_MODEL + c] = (y[c] - mu) * rs * lnw[c] + lnb[c];
}

// ---------------------------------------------------------------------------
extern "C" void kernel_launch(void* const* d_in, const int* in_sizes, int n_in,
                              void* d_out, int out_size, void* d_ws,
                              size_t ws_size, hipStream_t stream) {
  (void)in_sizes; (void)n_in; (void)out_size; (void)ws_size;
  const float* qx   = (const float*)d_in[0];
  const float* kx   = (const float*)d_in[1];
  const float* vx   = (const float*)d_in[2];
  const float* mask = (const float*)d_in[3];
  const float* Wq   = (const float*)d_in[4];
  const float* Wk   = (const float*)d_in[5];
  const float* Wv   = (const float*)d_in[6];
  const float* Wo   = (const float*)d_in[7];
  const float* sf   = (const float*)d_in[8];
  const float* tau  = (const float*)d_in[9];
  const float* lnw  = (const float*)d_in[10];
  const float* lnb  = (const float*)d_in[11];
  const int*   wsz  = (const int*)d_in[12];

  char* ws = (char*)d_ws;  // needs 56 MiB
  const size_t MB = (size_t)1024 * 1024;
  unsigned short* Xqbf = (unsigned short*)(ws);             // 8 MiB
  unsigned short* Xkbf = (unsigned short*)(ws + 8 * MB);    // 8 MiB
  unsigned short* Xvbf = (unsigned short*)(ws + 16 * MB);   // 8 MiB
  unsigned short* Wbf  = (unsigned short*)(ws + 24 * MB);   // 8 MiB (4 x 2)
  unsigned short* Qbf  = (unsigned short*)(ws + 32 * MB);   // 8 MiB
  unsigned short* Kbf  = (unsigned short*)(ws + 40 * MB);   // 8 MiB
  unsigned short* Vtbf = (unsigned short*)(ws + 48 * MB);   // 8 MiB
  unsigned short* Obf  = Xqbf;                              // reuse (dead)
  float*          Yf   = (float*)(ws + 8 * MB);             // reuse Xk+Xv

  float* out_x = (float*)d_out;
  float* att   = out_x + (size_t)ROWS * D_MODEL;

  zero_f4_kernel<<<4096, 256, 0, stream>>>(
      (float4*)att, (long long)B_SZ * H_N * L_SEQ * L_SEQ / 4);
  cvt_bf16_kernel<<<dim3(512, 7), 256, 0, stream>>>(
      qx, kx, vx, Wq, Wk, Wv, Wo,
      (unsigned*)Xqbf, (unsigned*)Xkbf, (unsigned*)Xvbf, (unsigned*)Wbf);
  proj_qkv_kernel<<<dim3(ROWS / 128, D_MODEL / 64, 3), 256, 0, stream>>>(
      Xqbf, Xkbf, Xvbf, Wbf, Qbf, Kbf, Vtbf);
  attn_kernel<<<(B_SZ * H_N * (L_SEQ / 16)) / 8, 256, 0, stream>>>(
      Qbf, Kbf, Vtbf, mask, sf, tau, wsz, att, Obf);
  out_proj_kernel<<<dim3(ROWS / 128, D_MODEL / 64), 256, 0, stream>>>(
      Obf, Wbf + (size_t)3 * D_MODEL * D_MODEL, qx, Yf);
  ln_kernel<<<ROWS, 256, 0, stream>>>(Yf, lnw, lnb, out_x);
}
